// TABGNNFused_74964359184698
// MI455X (gfx1250) — compile-verified, matrix operand order checked by reading, not run on hardware
//
#include <hip/hip_runtime.h>
#include <hip/hip_bf16.h>
#include <cstdint>
#include <cstddef>

// ---------------------------------------------------------------- types ----
typedef __bf16 bf16_t;
typedef bf16_t v16bf __attribute__((ext_vector_type(16)));
typedef bf16_t v4bf  __attribute__((ext_vector_type(4)));
typedef float  v8f   __attribute__((ext_vector_type(8)));

#define C_       128
#define NH_      8
#define S_       17
#define FUSED_   256
#define N_NODES  20000
#define N_EDGES  320000
#define EPS_     1e-5f

// ------------------------------------------------------------- helpers ----
__device__ __forceinline__ bf16_t f2bf(float f) {
  unsigned u = __builtin_bit_cast(unsigned, f);
  unsigned r = (u + 0x7FFFu + ((u >> 16) & 1u)) >> 16;   // RNE
  return __builtin_bit_cast(bf16_t, (unsigned short)r);
}
__device__ __forceinline__ v4bf f2bf4(float4 f) {
  v4bf o;
  o.x = f2bf(f.x); o.y = f2bf(f.y); o.z = f2bf(f.z); o.w = f2bf(f.w);
  return o;
}

// A-fragment from a bf16 row in LDS. rowp already includes the +8 element
// offset for lanes 16..31. Elements j<8 at rowp[0..7], j>=8 at rowp[16..23].
__device__ __forceinline__ v16bf afrag_from(const bf16_t* rowp) {
  union { v16bf v; float4 q[2]; } u;
  u.q[0] = *reinterpret_cast<const float4*>(rowp);
  u.q[1] = *reinterpret_cast<const float4*>(rowp + 16);
  return u.v;
}

// B-fragment from pre-swizzled weights: [ntile][ktile][lane][16] contiguous.
__device__ __forceinline__ v16bf bfrag_from(const bf16_t* W, int nt, int kt,
                                            int KT, int lane) {
  return *reinterpret_cast<const v16bf*>(
      W + ((((size_t)nt * KT + kt) * 32 + lane) << 4));
}

__device__ __forceinline__ v8f wmma_bf16(v16bf a, v16bf b, v8f c) {
  return __builtin_amdgcn_wmma_f32_16x16x32_bf16(
      /*neg_a=*/false, a, /*neg_b=*/false, b,
      /*c_mod=*/(short)0, c, /*reuse_a=*/false, /*reuse_b=*/false);
}

// order-preserving float<->uint keys for atomic max/min
__device__ __forceinline__ unsigned fkey(float f) {
  unsigned u = __builtin_bit_cast(unsigned, f);
  return (u & 0x80000000u) ? ~u : (u | 0x80000000u);
}
__device__ __forceinline__ float fromkey(unsigned k) {
  unsigned u = (k & 0x80000000u) ? (k & 0x7FFFFFFFu) : ~k;
  return __builtin_bit_cast(float, u);
}

// -------------------------------------------------- weight pack (f32->bf16,
// swizzled into per-wave WMMA B fragment order) ----------------------------
__global__ void k_pack(const float* __restrict__ W, bf16_t* __restrict__ out,
                       int N, int K, int total) {
  int idx = blockIdx.x * 256 + threadIdx.x;
  if (idx >= total) return;
  int j    = idx & 15;
  int lane = (idx >> 4) & 31;
  int rest = idx >> 9;
  int KT   = K >> 5;
  int kt   = rest % KT;
  int nt   = rest / KT;
  int n = nt * 16 + (lane & 15);
  int k = kt * 32 + ((lane & 16) ? 8 : 0) + ((j < 8) ? j : j + 8);
  out[idx] = f2bf(W[(size_t)n * K + k]);
}

// -------------------------------------------------------- generic GEMM ----
// Y[M,N] = act(A[M,K] @ W^T + bias). Block = 256 thr = 8 waves, tile 32x64.
// REQUIRES M % 32 == 0 (true for all launches: 20000, 320000).
// Double-buffered A tile: one barrier per K-step, loads overlap WMMA.
// grid.x = M/32, grid.y = N/64.  act: 0 none, 1 relu, 2 leaky(0.01)
__global__ __launch_bounds__(256) void k_gemm(
    const float* __restrict__ A, const bf16_t* __restrict__ W,
    const float* __restrict__ bias, float* __restrict__ Y,
    int M, int N, int K, int act) {
  __shared__ bf16_t As[2][32 * 32];
  const int KT = K >> 5;
  const int tid = threadIdx.x, lane = tid & 31, wv = tid >> 5;
  const int wm = wv & 1, wn = wv >> 1;
  const int bm = blockIdx.x * 32;
  const int ntile = blockIdx.y * 4 + wn;
  const int r = tid >> 3, kk = (tid & 7) * 4;   // per-thread float4 slot
  const float* Arow = A + (size_t)(bm + r) * K + kk;

  *reinterpret_cast<v4bf*>(&As[0][r * 32 + kk]) =
      f2bf4(*reinterpret_cast<const float4*>(Arow));
  __syncthreads();

  v8f acc = {0, 0, 0, 0, 0, 0, 0, 0};
  const bf16_t* rowp0 = &As[0][((wm << 4) + (lane & 15)) * 32 +
                              ((lane & 16) ? 8 : 0)];
  for (int kt = 0; kt < KT; ++kt) {
    float4 nv;
    if (kt + 1 < KT) {
      nv = *reinterpret_cast<const float4*>(Arow + (kt + 1) * 32);
      if (kt + 2 < KT)
        __builtin_prefetch(Arow + (kt + 2) * 32, 0, 1);  // global_prefetch_b8
    }
    acc = wmma_bf16(afrag_from(rowp0 + (kt & 1) * 1024),
                    bfrag_from(W, ntile, kt, KT, lane), acc);
    if (kt + 1 < KT)
      *reinterpret_cast<v4bf*>(&As[(kt + 1) & 1][r * 32 + kk]) = f2bf4(nv);
    __syncthreads();
  }
  int gn = ntile * 16 + (lane & 15);
  float bv = bias ? bias[gn] : 0.f;
  int mbase = bm + (wm << 4) + ((lane & 16) ? 8 : 0);
#pragma unroll
  for (int rr = 0; rr < 8; ++rr) {
    float v = acc[rr] + bv;
    if (act == 1) v = fmaxf(v, 0.f);
    else if (act == 2) v = (v > 0.f) ? v : 0.01f * v;
    Y[(size_t)(mbase + rr) * N + gn] = v;
  }
}

// -------------------------------------- PNA pre GEMM (gather + atomics) ---
// m[e] = pre_w @ [xg[col[e]], xg[row[e]], eh[e]] + b ; fused segment
// sum / sumsq / max / min scattered to per-node accumulators by col[e].
__global__ __launch_bounds__(256) void k_pre_gemm(
    const float* __restrict__ xg, const float* __restrict__ eh,
    const int* __restrict__ rowi, const int* __restrict__ coli,
    const bf16_t* __restrict__ W, const float* __restrict__ bias,
    float* accS, float* accQ, unsigned* accMx, unsigned* accMn) {
  __shared__ bf16_t As[32 * 32];
  const int tid = threadIdx.x, lane = tid & 31, wv = tid >> 5;
  const int wm = wv & 1, wn = wv >> 1;
  const int be = blockIdx.x * 32;
  const int ntile = blockIdx.y * 4 + wn;
  const int r = tid >> 3, kk = (tid & 7) * 4;
  const int e = be + r;
  const int ce = coli[e], re = rowi[e];   // hoisted out of K loop
  v8f acc = {0, 0, 0, 0, 0, 0, 0, 0};
  for (int kt = 0; kt < 12; ++kt) {  // K = 384
    __syncthreads();
    int k = kt * 32 + kk;            // 4-elem segment never crosses 128/256
    const float* src;
    if (k < 128)      src = &xg[(size_t)ce * 128 + k];
    else if (k < 256) src = &xg[(size_t)re * 128 + (k - 128)];
    else              src = &eh[(size_t)e * 128 + (k - 256)];
    *reinterpret_cast<v4bf*>(&As[r * 32 + kk]) =
        f2bf4(*reinterpret_cast<const float4*>(src));
    __syncthreads();
    const bf16_t* rowp =
        &As[((wm << 4) + (lane & 15)) * 32 + ((lane & 16) ? 8 : 0)];
    acc = wmma_bf16(afrag_from(rowp), bfrag_from(W, ntile, kt, 12, lane), acc);
  }
  int gn = ntile * 16 + (lane & 15);
  float bv = bias[gn];
  int ebase = be + (wm << 4) + ((lane & 16) ? 8 : 0);
#pragma unroll
  for (int rr = 0; rr < 8; ++rr) {
    int ee = ebase + rr;
    float v = acc[rr] + bv;
    size_t dst = (size_t)coli[ee] * 128 + gn;
    atomicAdd(accS + dst, v);
    atomicAdd(accQ + dst, v * v);
    atomicMax(accMx + dst, fkey(v));
    atomicMin(accMn + dst, fkey(v));
  }
}

// ------------------------------- PNA post GEMM (virtual 1664-wide input) --
__global__ __launch_bounds__(256) void k_post_gemm(
    const float* __restrict__ xg, const float* __restrict__ agg,
    const float* __restrict__ s1, const float* __restrict__ s2,
    const bf16_t* __restrict__ W, const float* __restrict__ bias,
    float* __restrict__ Y) {
  __shared__ bf16_t As[32 * 32];
  const int tid = threadIdx.x, lane = tid & 31, wv = tid >> 5;
  const int wm = wv & 1, wn = wv >> 1;
  const int bm = blockIdx.x * 32;
  const int ntile = blockIdx.y * 4 + wn;
  const int r = tid >> 3, kk = (tid & 7) * 4;
  const int gm = bm + r;
  const float sc1 = s1[gm], sc2 = s2[gm];   // hoisted
  v8f acc = {0, 0, 0, 0, 0, 0, 0, 0};
  for (int kt = 0; kt < 52; ++kt) {  // K = 1664
    __syncthreads();
    int k = kt * 32 + kk;
    float4 v;
    if (k < 128) {
      v = *reinterpret_cast<const float4*>(&xg[(size_t)gm * 128 + k]);
    } else {
      int k2 = k - 128;
      int blk = k2 >> 9, id = k2 & 511;   // segment stays inside one block
      v = *reinterpret_cast<const float4*>(&agg[(size_t)gm * 512 + id]);
      float s = (blk == 1) ? sc1 : (blk == 2) ? sc2 : 1.f;
      v.x *= s; v.y *= s; v.z *= s; v.w *= s;
    }
    *reinterpret_cast<v4bf*>(&As[r * 32 + kk]) = f2bf4(v);
    __syncthreads();
    const bf16_t* rowp =
        &As[((wm << 4) + (lane & 15)) * 32 + ((lane & 16) ? 8 : 0)];
    acc = wmma_bf16(afrag_from(rowp), bfrag_from(W, ntile, kt, 52, lane), acc);
  }
  int gn = ntile * 16 + (lane & 15);
  float bv = bias[gn];
  int mbase = bm + (wm << 4) + ((lane & 16) ? 8 : 0);
#pragma unroll
  for (int rr = 0; rr < 8; ++rr)
    Y[(size_t)(mbase + rr) * 128 + gn] = acc[rr] + bv;
}

// -------------------------------------- fused transformer layer (1 node) --
// In-LDS tile GEMM: rows 0..16 valid (padded to 32), K = 128 (KT = 4).
__device__ __forceinline__ void tl_gemm(const bf16_t* ab, const bf16_t* W,
                                        const float* bias, float* dst, int NT,
                                        int wv, int lane, int act,
                                        int dstStride) {
  for (int t = wv; t < 2 * NT; t += 4) {
    int wm = t & 1, nt = t >> 1;
    v8f acc = {0, 0, 0, 0, 0, 0, 0, 0};
#pragma unroll
    for (int kt = 0; kt < 4; ++kt) {
      const bf16_t* rowp =
          &ab[((wm << 4) + (lane & 15)) * C_ + kt * 32 + ((lane & 16) ? 8 : 0)];
      acc = wmma_bf16(afrag_from(rowp), bfrag_from(W, nt, kt, 4, lane), acc);
    }
    int gn = nt * 16 + (lane & 15);
    float bv = bias[gn];
    int mb = (wm << 4) + ((lane & 16) ? 8 : 0);
#pragma unroll
    for (int r = 0; r < 8; ++r) {
      int gm = mb + r;
      if (gm < S_) {
        float v = acc[r] + bv;
        if (act) v = fmaxf(v, 0.f);
        dst[gm * dstStride + gn] = v;
      }
    }
  }
}

__device__ __forceinline__ void ln_row_ip(float* x, const float* w,
                                          const float* b) {
  float mu = 0.f;
  for (int i = 0; i < C_; ++i) mu += x[i];
  mu *= (1.f / C_);
  float var = 0.f;
  for (int i = 0; i < C_; ++i) { float d = x[i] - mu; var += d * d; }
  var *= (1.f / C_);
  float iv = rsqrtf(var + EPS_);
  for (int i = 0; i < C_; ++i) x[i] = (x[i] - mu) * iv * w[i] + b[i];
}

// mode 0: xt <- LN_tn(TL(xt));  mode 1: xt <- xt + LN_tn(TL(xt))/2
__global__ __launch_bounds__(128) void k_tl(
    float* __restrict__ xt, const bf16_t* Wq, const float* in_b,
    const bf16_t* Wo, const float* out_b, const bf16_t* W1, const float* b1,
    const bf16_t* W2, const float* b2, const float* n1w, const float* n1b,
    const float* n2w, const float* n2b, const float* tnw, const float* tnb,
    int mode) {
  __shared__ float  xin[S_ * C_];
  __shared__ float  xres[S_ * C_];
  __shared__ bf16_t ab[32 * C_];
  __shared__ float  tmp[S_ * 384];
  const int tid = threadIdx.x, lane = tid & 31, wv = tid >> 5;
  float* base = xt + (size_t)blockIdx.x * (S_ * C_);

  for (int i4 = tid * 4; i4 < 32 * C_; i4 += 512) {
    int r = i4 >> 7, c = i4 & 127;
    float4 v = make_float4(0.f, 0.f, 0.f, 0.f);
    if (r < S_) {
      v = *reinterpret_cast<const float4*>(&base[r * C_ + c]);
      *reinterpret_cast<float4*>(&xin[r * C_ + c]) = v;
    }
    *reinterpret_cast<v4bf*>(&ab[i4]) = f2bf4(v);   // rows 17..31 stay zero
  }
  __syncthreads();
  tl_gemm(ab, Wq, in_b, tmp, 24, wv, lane, 0, 384);   // QKV: 17x384
  __syncthreads();
  // attention: 8 heads x 17 query rows on VALU
  for (int task = tid; task < NH_ * S_; task += 128) {
    int q = task % S_, h = task / S_;
    const float* qr = &tmp[q * 384 + h * 16];
    float s[S_];
    float mx = -1e30f;
    for (int k = 0; k < S_; ++k) {
      const float* kr = &tmp[k * 384 + 128 + h * 16];
      float d = 0.f;
#pragma unroll
      for (int i = 0; i < 16; ++i) d += qr[i] * kr[i];
      d *= 0.25f;                                  // 1/sqrt(16)
      s[k] = d;
      mx = fmaxf(mx, d);
    }
    float sum = 0.f;
    for (int k = 0; k < S_; ++k) { float e = __expf(s[k] - mx); s[k] = e; sum += e; }
    float inv = 1.f / sum;
    for (int d = 0; d < 16; ++d) {
      float o = 0.f;
      for (int k = 0; k < S_; ++k) o += s[k] * tmp[k * 384 + 256 + h * 16 + d];
      xres[q * C_ + h * 16 + d] = o * inv;
    }
  }
  __syncthreads();
  for (int i4 = tid * 4; i4 < S_ * C_; i4 += 512)
    *reinterpret_cast<v4bf*>(&ab[i4]) =
        f2bf4(*reinterpret_cast<const float4*>(&xres[i4]));
  __syncthreads();
  tl_gemm(ab, Wo, out_b, tmp, 8, wv, lane, 0, C_);    // out proj
  __syncthreads();
  for (int i4 = tid * 4; i4 < S_ * C_; i4 += 512) {
    float4 a = *reinterpret_cast<const float4*>(&xin[i4]);
    float4 b = *reinterpret_cast<const float4*>(&tmp[i4]);
    a.x += b.x; a.y += b.y; a.z += b.z; a.w += b.w;
    *reinterpret_cast<float4*>(&xres[i4]) = a;
  }
  __syncthreads();
  if (tid < S_) {                                     // LN1
    float* x = &xres[tid * C_];
    ln_row_ip(x, n1w, n1b);
    for (int i = 0; i < C_; ++i) ab[tid * C_ + i] = f2bf(x[i]);
  }
  __syncthreads();
  tl_gemm(ab, W1, b1, tmp, 8, wv, lane, 1, C_);       // FF1 + relu
  __syncthreads();
  for (int i4 = tid * 4; i4 < S_ * C_; i4 += 512)
    *reinterpret_cast<v4bf*>(&ab[i4]) =
        f2bf4(*reinterpret_cast<const float4*>(&tmp[i4]));
  __syncthreads();
  tl_gemm(ab, W2, b2, tmp + S_ * C_, 8, wv, lane, 0, C_);  // FF2
  __syncthreads();
  for (int i4 = tid * 4; i4 < S_ * C_; i4 += 512) {
    float4 a = *reinterpret_cast<const float4*>(&xres[i4]);
    float4 b = *reinterpret_cast<const float4*>(&tmp[S_ * C_ + i4]);
    a.x += b.x; a.y += b.y; a.z += b.z; a.w += b.w;
    *reinterpret_cast<float4*>(&xres[i4]) = a;
  }
  __syncthreads();
  if (tid < S_) ln_row_ip(&xres[tid * C_], n2w, n2b);      // LN2
  __syncthreads();
  if (tid < S_) {                                          // outer LN + mode
    float* x = &xres[tid * C_];
    float mu = 0.f;
    for (int i = 0; i < C_; ++i) mu += x[i];
    mu *= (1.f / C_);
    float var = 0.f;
    for (int i = 0; i < C_; ++i) { float d = x[i] - mu; var += d * d; }
    var *= (1.f / C_);
    float iv = rsqrtf(var + EPS_);
    for (int i = 0; i < C_; ++i) {
      float v = (x[i] - mu) * iv * tnw[i] + tnb[i];
      base[tid * C_ + i] = mode ? (xin[tid * C_ + i] + 0.5f * v) : v;
    }
  }
}

// ------------------------------------------------------ small kernels -----
__global__ void k_build_xt(const float* __restrict__ x,
                           const float* __restrict__ cls,
                           float* __restrict__ xt, int total) {
  int idx = blockIdx.x * 256 + threadIdx.x;
  if (idx >= total) return;
  int c = idx % 128, s = (idx / 128) % 17, b = idx / (17 * 128);
  xt[idx] = (s == 0) ? cls[c] : x[(size_t)b * 2048 + (s - 1) * 128 + c];
}

__global__ void k_deg(const int* __restrict__ coli, float* deg) {
  int e = blockIdx.x * 256 + threadIdx.x;
  if (e < N_EDGES) atomicAdd(&deg[coli[e]], 1.f);
}

__global__ void k_avglog(const float* __restrict__ deg, float* acc) {
  int n = blockIdx.x * 256 + threadIdx.x;
  if (n < N_NODES) atomicAdd(acc, logf(deg[n] + 1.f));
}

__global__ void k_combine(const float* accS, const float* accQ,
                          const unsigned* accMx, const unsigned* accMn,
                          const float* deg, const float* avgsum, float* agg,
                          float* s1, float* s2, int total) {
  int idx = blockIdx.x * 256 + threadIdx.x;
  if (idx >= total) return;
  int c = idx & 127, n = idx >> 7;
  float dv = deg[n];
  float dc = fmaxf(dv, 1.f);
  float mean = accS[idx] / dc;
  float m2 = accQ[idx] / dc;
  float sd = sqrtf(fmaxf(m2 - mean * mean, 0.f) + EPS_);
  bool has = dv > 0.f;
  float mx = has ? fromkey(accMx[idx]) : 0.f;
  float mn = has ? fromkey(accMn[idx]) : 0.f;
  float* a = agg + (size_t)n * 512;
  a[c] = mean; a[128 + c] = mx; a[256 + c] = mn; a[384 + c] = sd;
  if (c == 0) {
    float ld = logf(dc + 1.f);
    float av = avgsum[0] * (1.f / N_NODES);
    s1[n] = ld / av;
    s2[n] = av / ld;
  }
}

__global__ void k_bn_stats(const float* __restrict__ g, float* bs, float* bq,
                           int total) {
  int idx = blockIdx.x * 256 + threadIdx.x;
  if (idx >= total) return;
  int c = idx & 127;
  float v = g[idx];
  atomicAdd(&bs[c], v);
  atomicAdd(&bq[c], v * v);
}

__global__ void k_bn_apply(const float* __restrict__ xg,
                           const float* __restrict__ g, const float* gam,
                           const float* bet, const float* bs, const float* bq,
                           float* __restrict__ out, int total) {
  int idx = blockIdx.x * 256 + threadIdx.x;
  if (idx >= total) return;
  int c = idx & 127;
  float mean = bs[c] * (1.f / N_NODES);
  float var = bq[c] * (1.f / N_NODES) - mean * mean;
  float v = (g[idx] - mean) * rsqrtf(var + EPS_) * gam[c] + bet[c];
  out[idx] = (xg[idx] + fmaxf(v, 0.f)) * 0.5f;
}

__global__ __launch_bounds__(256) void k_concat_ln(
    const float* __restrict__ xt, const float* __restrict__ xg,
    const float* w, const float* b, float* __restrict__ out) {
  __shared__ float red[256];
  int n = blockIdx.x, c = threadIdx.x;
  float v = (c < 128) ? xt[(size_t)n * (17 * 128) + c]
                      : xg[(size_t)n * 128 + (c - 128)];
  red[c] = v;
  __syncthreads();
  for (int s = 128; s > 0; s >>= 1) { if (c < s) red[c] += red[c + s]; __syncthreads(); }
  float mu = red[0] * (1.f / 256.f);
  __syncthreads();
  float d = v - mu;
  red[c] = d * d;
  __syncthreads();
  for (int s = 128; s > 0; s >>= 1) { if (c < s) red[c] += red[c + s]; __syncthreads(); }
  float iv = rsqrtf(red[0] * (1.f / 256.f) + EPS_);
  out[(size_t)n * 256 + c] = (v - mu) * iv * w[c] + b[c];
}

__global__ __launch_bounds__(256) void k_ln256(const float* __restrict__ X,
                                               const float* w, const float* b,
                                               float* __restrict__ out) {
  __shared__ float red[256];
  int n = blockIdx.x, c = threadIdx.x;
  float v = X[(size_t)n * 256 + c];
  red[c] = v;
  __syncthreads();
  for (int s = 128; s > 0; s >>= 1) { if (c < s) red[c] += red[c + s]; __syncthreads(); }
  float mu = red[0] * (1.f / 256.f);
  __syncthreads();
  float d = v - mu;
  red[c] = d * d;
  __syncthreads();
  for (int s = 128; s > 0; s >>= 1) { if (c < s) red[c] += red[c + s]; __syncthreads(); }
  float iv = rsqrtf(red[0] * (1.f / 256.f) + EPS_);
  out[(size_t)n * 256 + c] = (v - mu) * iv * w[c] + b[c];
}

__global__ void k_resid(float* __restrict__ xt, const float* __restrict__ f,
                        const float* __restrict__ xgB, float* __restrict__ xgA,
                        int total) {
  int idx = blockIdx.x * 256 + threadIdx.x;
  if (idx >= total) return;
  int c = idx & 127, n = idx >> 7;
  size_t xo = (size_t)n * (17 * 128) + c;
  xt[xo] = (xt[xo] + f[(size_t)n * 256 + c]) * 0.5f;
  xgA[idx] = (xgB[idx] + f[(size_t)n * 256 + 128 + c]) * 0.5f;
}

// ---------------------------------------------------------------- host ----
static void* bump(char*& p, size_t bytes) {
  void* r = (void*)p;
  p += (bytes + 255) & ~(size_t)255;
  return r;
}

extern "C" void kernel_launch(void* const* d_in, const int* in_sizes, int n_in,
                              void* d_out, int out_size, void* d_ws,
                              size_t ws_size, hipStream_t stream) {
  (void)in_sizes; (void)n_in; (void)out_size; (void)ws_size;
  auto F = [&](int i) { return (const float*)d_in[i]; };
  const float* x         = F(0);
  const int*   ei        = (const int*)d_in[1];
  const float* edge_attr = F(2);
  const float* cls       = F(5);
  const int* rowi = ei;
  const int* coli = ei + N_EDGES;

  // ---- workspace layout ----
  char* wp = (char*)d_ws;
  float*    xt     = (float*)bump(wp, (size_t)N_NODES * 17 * 128 * 4);
  float*    eh     = (float*)bump(wp, (size_t)N_EDGES * 128 * 4);  // aliases f1/f2
  float*    accS   = (float*)bump(wp, (size_t)N_NODES * 128 * 4);
  float*    accQ   = (float*)bump(wp, (size_t)N_NODES * 128 * 4);
  unsigned* accMx  = (unsigned*)bump(wp, (size_t)N_NODES * 128 * 4);
  unsigned* accMn  = (unsigned*)bump(wp, (size_t)N_NODES * 128 * 4);
  float*    agg    = (float*)bump(wp, (size_t)N_NODES * 512 * 4);
  float*    s1     = (float*)bump(wp, (size_t)N_NODES * 4);
  float*    s2     = (float*)bump(wp, (size_t)N_NODES * 4);
  float*    deg    = (float*)bump(wp, (size_t)N_NODES * 4);
  float*    avgsum = (float*)bump(wp, 256);
  float*    g1     = (float*)bump(wp, (size_t)N_NODES * 128 * 4);
  float*    g      = (float*)bump(wp, (size_t)N_NODES * 128 * 4);
  float*    bnS    = (float*)bump(wp, 128 * 4);
  float*    bnQ    = (float*)bump(wp, 128 * 4);
  float*    xgA    = (float*)bump(wp, (size_t)N_NODES * 128 * 4);
  float*    xgB    = (float*)bump(wp, (size_t)N_NODES * 128 * 4);
  float*    fbuf   = (float*)bump(wp, (size_t)N_NODES * 256 * 4);
  float*    f3     = (float*)bump(wp, (size_t)N_NODES * 256 * 4);
  bf16_t*   arena  = (bf16_t*)bump(wp, (size_t)4400000 * 2);
  float*    f1 = eh;                           // dead eh reused: 2 x 82 MB
  float*    f2 = eh + (size_t)N_NODES * 1024;
  float*    eaOut = (float*)d_out + (size_t)N_NODES * 128;

  // ---- pack all weights to bf16 fragment order ----
  size_t aoff = 0;
  auto pack = [&](const float* W, int N, int K) -> bf16_t* {
    bf16_t* dst = arena + aoff;
    aoff += (size_t)N * K;
    int total = N * K;
    k_pack<<<dim3((total + 255) / 256), dim3(256), 0, stream>>>(W, dst, N, K, total);
    return dst;
  };
  bf16_t* Wq0   = pack(F(10), 384, 128);
  bf16_t* Wo0   = pack(F(12), 128, 128);
  bf16_t* W10   = pack(F(14), 128, 128);
  bf16_t* W20   = pack(F(16), 128, 128);
  bf16_t* Wnode = pack(F(6), 128, 2048);
  bf16_t* Wedge = pack(F(8), 128, 64);
  struct LW { bf16_t *Wq, *Wo, *W1, *W2, *Wee, *Wpre, *Wpost, *Wlin, *Wf1, *Wf2, *Wf3; } lw[2];
  for (int l = 0; l < 2; ++l) {
    int b0 = 24 + l * 34;
    lw[l].Wq    = pack(F(b0 + 0), 384, 128);
    lw[l].Wo    = pack(F(b0 + 2), 128, 128);
    lw[l].W1    = pack(F(b0 + 4), 128, 128);
    lw[l].W2    = pack(F(b0 + 6), 128, 128);
    lw[l].Wee   = pack(F(b0 + 14), 128, 128);
    lw[l].Wpre  = pack(F(b0 + 16), 128, 384);
    lw[l].Wpost = pack(F(b0 + 18), 128, 1664);
    lw[l].Wlin  = pack(F(b0 + 20), 128, 128);
    lw[l].Wf1   = pack(F(b0 + 26), 1024, 256);
    lw[l].Wf2   = pack(F(b0 + 28), 1024, 1024);
    lw[l].Wf3   = pack(F(b0 + 30), 256, 1024);
  }

  // ---- ea = edge_attr @ edge_w^T + b (returned output, used by PNA) ----
  k_gemm<<<dim3(N_EDGES / 32, 2), 256, 0, stream>>>(edge_attr, Wedge, F(9),
                                                    eaOut, N_EDGES, 128, 64, 0);
  // ---- x_gnn0 = x.reshape(N,2048) @ node_w^T + b ----
  k_gemm<<<dim3(N_NODES / 32, 2), 256, 0, stream>>>(x, Wnode, F(7), xgA,
                                                    N_NODES, 128, 2048, 0);
  // ---- xt = [cls ; x] ; tab0 transformer + tn0 LN ----
  int totxt = N_NODES * 17 * 128;
  k_build_xt<<<dim3((totxt + 255) / 256), dim3(256), 0, stream>>>(x, cls, xt, totxt);
  k_tl<<<dim3(N_NODES), dim3(128), 0, stream>>>(
      xt, Wq0, F(11), Wo0, F(13), W10, F(15), W20, F(17), F(18), F(19), F(20),
      F(21), F(22), F(23), 0);

  // ---- degree & avg_log ----
  hipMemsetAsync(deg, 0, N_NODES * 4, stream);
  hipMemsetAsync(avgsum, 0, 4, stream);
  k_deg<<<dim3((N_EDGES + 255) / 256), dim3(256), 0, stream>>>(coli, deg);
  k_avglog<<<dim3((N_NODES + 255) / 256), dim3(256), 0, stream>>>(deg, avgsum);

  const size_t accB = (size_t)N_NODES * 128 * 4;
  const int totNC = N_NODES * 128;
  for (int l = 0; l < 2; ++l) {
    int b0 = 24 + l * 34;
    // xt = xt + LN(TL(xt))/2
    k_tl<<<dim3(N_NODES), dim3(128), 0, stream>>>(
        xt, lw[l].Wq, F(b0 + 1), lw[l].Wo, F(b0 + 3), lw[l].W1, F(b0 + 5),
        lw[l].W2, F(b0 + 7), F(b0 + 8), F(b0 + 9), F(b0 + 10), F(b0 + 11),
        F(b0 + 12), F(b0 + 13), 1);
    // eh = ea @ ee_w^T + b
    k_gemm<<<dim3(N_EDGES / 32, 2), 256, 0, stream>>>(eaOut, lw[l].Wee,
                                                      F(b0 + 15), eh, N_EDGES,
                                                      128, 128, 0);
    // PNA aggregation
    hipMemsetAsync(accS, 0, accB, stream);
    hipMemsetAsync(accQ, 0, accB, stream);
    hipMemsetAsync(accMx, 0, accB, stream);
    hipMemsetAsync(accMn, 0xFF, accB, stream);
    k_pre_gemm<<<dim3(N_EDGES / 32, 2), 256, 0, stream>>>(
        xgA, eh, rowi, coli, lw[l].Wpre, F(b0 + 17), accS, accQ, accMx, accMn);
    k_combine<<<dim3((totNC + 255) / 256), dim3(256), 0, stream>>>(
        accS, accQ, accMx, accMn, deg, avgsum, agg, s1, s2, totNC);
    k_post_gemm<<<dim3(N_NODES / 32, 2), 256, 0, stream>>>(
        xgA, agg, s1, s2, lw[l].Wpost, F(b0 + 19), g1);
    k_gemm<<<dim3(N_NODES / 32, 2), 256, 0, stream>>>(g1, lw[l].Wlin,
                                                      F(b0 + 21), g, N_NODES,
                                                      128, 128, 0);
    // batchnorm + relu + residual -> xgB
    hipMemsetAsync(bnS, 0, 128 * 4, stream);
    hipMemsetAsync(bnQ, 0, 128 * 4, stream);
    k_bn_stats<<<dim3((totNC + 255) / 256), dim3(256), 0, stream>>>(g, bnS, bnQ, totNC);
    k_bn_apply<<<dim3((totNC + 255) / 256), dim3(256), 0, stream>>>(
        xgA, g, F(b0 + 22), F(b0 + 23), bnS, bnQ, xgB, totNC);
    // fused MLP over [cls_t ; x_gnn]
    k_concat_ln<<<dim3(N_NODES), dim3(256), 0, stream>>>(xt, xgB, F(b0 + 24),
                                                         F(b0 + 25), fbuf);
    k_gemm<<<dim3(N_NODES / 32, 16), 256, 0, stream>>>(fbuf, lw[l].Wf1,
                                                       F(b0 + 27), f1, N_NODES,
                                                       1024, 256, 2);
    k_gemm<<<dim3(N_NODES / 32, 16), 256, 0, stream>>>(f1, lw[l].Wf2,
                                                       F(b0 + 29), f2, N_NODES,
                                                       1024, 1024, 2);
    k_gemm<<<dim3(N_NODES / 32, 4), 256, 0, stream>>>(f2, lw[l].Wf3,
                                                      F(b0 + 31), f3, N_NODES,
                                                      256, 1024, 0);
    k_ln256<<<dim3(N_NODES), dim3(256), 0, stream>>>(f3, F(b0 + 32), F(b0 + 33), fbuf);
    // cls_t, x_gnn residual halves -> xt row0, xgA
    k_resid<<<dim3((totNC + 255) / 256), dim3(256), 0, stream>>>(xt, fbuf, xgB,
                                                                 xgA, totNC);
  }
  hipMemcpyAsync(d_out, xgA, (size_t)N_NODES * 128 * 4,
                 hipMemcpyDeviceToDevice, stream);
}
